// NN_78331613544881
// MI455X (gfx1250) — matrix operand order repaired and checked by reading, not compiled, not run on hardware
//
#include <hip/hip_runtime.h>
#include <hip/hip_bf16.h>

#define B_ROWS 4096
#define NFEAT  41920
#define HACC   256
#define NVEC4  (NFEAT / 4)   // 10480

typedef __attribute__((ext_vector_type(16))) _Float16 v16h;
typedef __attribute__((ext_vector_type(8)))  _Float16 v8h;
typedef __attribute__((ext_vector_type(8)))  float    v8f;
typedef __attribute__((ext_vector_type(4)))  float    v4f;

// ---------------------------------------------------------------------------
// Kernel 1: convert W1 [32,512] and W2 [32,32] fp32 -> f16 in workspace
// ---------------------------------------------------------------------------
__global__ void nnue_convert_w(const float* __restrict__ W1,
                               const float* __restrict__ W2,
                               _Float16* __restrict__ W1h,
                               _Float16* __restrict__ W2h) {
    int i = blockIdx.x * blockDim.x + threadIdx.x;
    if (i < 32 * 512) {
        W1h[i] = (_Float16)W1[i];
    } else {
        int j = i - 32 * 512;
        if (j < 32 * 32) W2h[j] = (_Float16)W2[j];
    }
}

// ---------------------------------------------------------------------------
// Kernel 2: tiled transpose Ww/Wb [256, F] -> [F, 256] (coalesced gathers later)
// grid (F/32, 256/32, 2), block (32, 8)
// ---------------------------------------------------------------------------
__global__ void nnue_transpose_w(const float* __restrict__ Ww,
                                 const float* __restrict__ Wb,
                                 float* __restrict__ WwT,
                                 float* __restrict__ WbT) {
    __shared__ float tile[32][33];
    const float* src = blockIdx.z ? Wb : Ww;
    float*       dst = blockIdx.z ? WbT : WwT;
    const int f0 = blockIdx.x * 32;
    const int h0 = blockIdx.y * 32;
    const int tx = threadIdx.x, ty = threadIdx.y;
#pragma unroll
    for (int i = 0; i < 32; i += 8)
        tile[ty + i][tx] = src[(size_t)(h0 + ty + i) * NFEAT + (f0 + tx)];
    __syncthreads();
#pragma unroll
    for (int i = 0; i < 32; i += 8)
        dst[(size_t)(f0 + ty + i) * HACC + (h0 + tx)] = tile[tx][ty + i];
}

// ---------------------------------------------------------------------------
// Kernel 3: per-row sparse scan + accumulate + stm select + clip -> f16 acts
// One 256-thread block (8 waves) per batch row. Thread tid owns h = tid.
// Feature stream (1.37 GB, zero reuse) uses NT loads so it does not evict
// the L2-resident transposed weights (86 MB of MI455X's 192 MB L2).
// ---------------------------------------------------------------------------
__global__ __launch_bounds__(256) void nnue_row_kernel(
    const float* __restrict__ wfeat, const float* __restrict__ bfeat,
    const float* __restrict__ stm,
    const float* __restrict__ Wwg, const float* __restrict__ bw,
    const float* __restrict__ Wbg, const float* __restrict__ bb,
    _Float16* __restrict__ act, int transposed) {
    __shared__ int s_cnt[2];
    __shared__ int s_idx[2][64];
    const int b   = blockIdx.x;
    const int tid = threadIdx.x;
    if (tid < 2) s_cnt[tid] = 0;
    __syncthreads();

    const v4f* wrow = (const v4f*)(wfeat + (size_t)b * NFEAT);
    const v4f* brow = (const v4f*)(bfeat + (size_t)b * NFEAT);
    for (int i = tid; i < NVEC4; i += 256) {
        v4f v = __builtin_nontemporal_load(wrow + i);   // TH=NT global_load_b128
        if (v[0] != 0.0f || v[1] != 0.0f || v[2] != 0.0f || v[3] != 0.0f) {
#pragma unroll
            for (int c = 0; c < 4; ++c)
                if (v[c] != 0.0f) {
                    int p = atomicAdd(&s_cnt[0], 1);
                    if (p < 64) s_idx[0][p] = 4 * i + c;
                }
        }
        v4f u = __builtin_nontemporal_load(brow + i);
        if (u[0] != 0.0f || u[1] != 0.0f || u[2] != 0.0f || u[3] != 0.0f) {
#pragma unroll
            for (int c = 0; c < 4; ++c)
                if (u[c] != 0.0f) {
                    int p = atomicAdd(&s_cnt[1], 1);
                    if (p < 64) s_idx[1][p] = 4 * i + c;
                }
        }
    }
    __syncthreads();

    int nW = min(s_cnt[0], 64);
    int nB = min(s_cnt[1], 64);
    float accW = bw[tid];
    float accB = bb[tid];
    if (transposed) {  // coalesced: WT[f*256 + h], L2-resident
        for (int j = 0; j < nW; ++j) accW += Wwg[(size_t)s_idx[0][j] * HACC + tid];
        for (int j = 0; j < nB; ++j) accB += Wbg[(size_t)s_idx[1][j] * HACC + tid];
    } else {           // strided fallback: W[h*F + f]
        for (int j = 0; j < nW; ++j) accW += Wwg[(size_t)tid * NFEAT + s_idx[0][j]];
        for (int j = 0; j < nB; ++j) accB += Wbg[(size_t)tid * NFEAT + s_idx[1][j]];
    }

    const float s = stm[b];                 // 0.0 or 1.0
    float first  = accW + s * (accB - accW);
    float second = accB + s * (accW - accB);
    first  = fminf(fmaxf(first,  0.0f), 1.0f);
    second = fminf(fmaxf(second, 0.0f), 1.0f);
    act[(size_t)b * 512 + tid]       = (_Float16)first;
    act[(size_t)b * 512 + 256 + tid] = (_Float16)second;
}

// ---------------------------------------------------------------------------
// Kernel 4: MLP via v_wmma_f32_16x16x32_f16. One wave per 16-row tile.
//   h1 = clip(act @ W1^T + b1); h2 = clip(h1 @ W2^T + b2); out = h2 @ Wo^T + bo
// A layout (16-bit 16x32): lane l: M=l&15, kg=(l>>4)*16, K in {kg..kg+15} (contiguous)
// B layout (16-bit 32x16): lane l: N=l&15, kb=(l>>4)*16, K in {kb..kb+15} (contiguous)
// C/D layout: VGPR r, lane l -> M = r + 8*(l>=16), N = l&15
// ---------------------------------------------------------------------------
__device__ inline v16h load_a_tile(const _Float16* p) {
    v8h lo = *(const v8h*)p;
    v8h hi = *(const v8h*)(p + 8);
    v16h r;
#pragma unroll
    for (int i = 0; i < 8; ++i) { r[i] = lo[i]; r[i + 8] = hi[i]; }
    return r;
}

__global__ __launch_bounds__(32) void nnue_mlp_wmma(
    const _Float16* __restrict__ act,
    const _Float16* __restrict__ W1h, const _Float16* __restrict__ W2h,
    const float* __restrict__ b1, const float* __restrict__ b2,
    const float* __restrict__ Wo, const float* __restrict__ bo,
    float* __restrict__ out) {
    __shared__ _Float16 s_h1[16 * 32];
    __shared__ float    s_h2[16 * 32];
    const int tid     = threadIdx.x;
    const int rowbase = blockIdx.x * 16;
    const int m  = tid & 15;          // A row / B column (within tile)
    const int kg = (tid >> 4) * 16;   // A-side K group base
    const int kb = (tid >> 4) * 16;   // B-side K group base
    const int Mb = (tid >> 4) * 8;    // C/D M base

    // ---- layer 1: [16,512] x [512,32] ----
    v8f c0 = {};
    v8f c1 = {};
    const _Float16* arow  = act + (size_t)(rowbase + m) * 512;
    const _Float16* w1r0  = W1h + (size_t)m * 512;         // N-tile 0: W1 rows 0..15
    const _Float16* w1r1  = W1h + (size_t)(16 + m) * 512;  // N-tile 1: W1 rows 16..31
#pragma unroll 4
    for (int kk = 0; kk < 16; ++kk) {
        const int k0 = kk * 32;
        v16h a   = load_a_tile(arow + k0 + kg);
        v16h bm0 = *(const v16h*)(w1r0 + k0 + kb);
        v16h bm1 = *(const v16h*)(w1r1 + k0 + kb);
        c0 = __builtin_amdgcn_wmma_f32_16x16x32_f16(false, a, false, bm0, (short)0, c0, false, false);
        c1 = __builtin_amdgcn_wmma_f32_16x16x32_f16(false, a, false, bm1, (short)0, c1, false, false);
    }
    const float b1A = b1[m], b1B = b1[16 + m];
#pragma unroll
    for (int r = 0; r < 8; ++r) {
        float v = c0[r] + b1A; v = fminf(fmaxf(v, 0.0f), 1.0f);
        float w = c1[r] + b1B; w = fminf(fmaxf(w, 0.0f), 1.0f);
        s_h1[(Mb + r) * 32 + m]      = (_Float16)v;
        s_h1[(Mb + r) * 32 + 16 + m] = (_Float16)w;
    }
    __syncthreads();

    // ---- layer 2: [16,32] x [32,32], single K-step ----
    v16h a2  = load_a_tile(&s_h1[m * 32 + kg]);
    v16h w2a = *(const v16h*)(W2h + (size_t)m * 32 + kb);
    v16h w2b = *(const v16h*)(W2h + (size_t)(16 + m) * 32 + kb);
    v8f d0 = {};
    v8f d1 = {};
    d0 = __builtin_amdgcn_wmma_f32_16x16x32_f16(false, a2, false, w2a, (short)0, d0, false, false);
    d1 = __builtin_amdgcn_wmma_f32_16x16x32_f16(false, a2, false, w2b, (short)0, d1, false, false);
    const float b2A = b2[m], b2B = b2[16 + m];
#pragma unroll
    for (int r = 0; r < 8; ++r) {
        float v = d0[r] + b2A; v = fminf(fmaxf(v, 0.0f), 1.0f);
        float w = d1[r] + b2B; w = fminf(fmaxf(w, 0.0f), 1.0f);
        s_h2[(Mb + r) * 32 + m]      = v;
        s_h2[(Mb + r) * 32 + 16 + m] = w;
    }
    __syncthreads();

    // ---- output layer: dot(h2[m], Wo) + bo ----
    if (tid < 16) {
        float accv = bo[0];
#pragma unroll
        for (int n = 0; n < 32; ++n) accv += s_h2[tid * 32 + n] * Wo[n];
        out[rowbase + tid] = accv;
    }
}

// ---------------------------------------------------------------------------
extern "C" void kernel_launch(void* const* d_in, const int* in_sizes, int n_in,
                              void* d_out, int out_size, void* d_ws, size_t ws_size,
                              hipStream_t stream) {
    const float* wfeat = (const float*)d_in[0];
    const float* bfeat = (const float*)d_in[1];
    const float* stm   = (const float*)d_in[2];
    const float* Ww    = (const float*)d_in[3];
    const float* bw    = (const float*)d_in[4];
    const float* Wb    = (const float*)d_in[5];
    const float* bb    = (const float*)d_in[6];
    const float* W1    = (const float*)d_in[7];
    const float* b1    = (const float*)d_in[8];
    const float* W2    = (const float*)d_in[9];
    const float* b2    = (const float*)d_in[10];
    const float* Wo    = (const float*)d_in[11];
    const float* bo    = (const float*)d_in[12];
    float* out = (float*)d_out;

    // workspace layout
    const size_t ACT_OFF = 0;                                   // [4096,512] f16 = 4 MiB
    const size_t W1H_OFF = (size_t)B_ROWS * 512 * 2;            // 4,194,304
    const size_t W2H_OFF = W1H_OFF + 32 * 512 * 2;              // +32 KiB
    const size_t WT_OFF  = ((W2H_OFF + 32 * 32 * 2 + 255) / 256) * 256;
    const size_t WT_BYTES = (size_t)NFEAT * HACC * sizeof(float);   // 42,926,080 each

    char* ws = (char*)d_ws;
    _Float16* act = (_Float16*)(ws + ACT_OFF);
    _Float16* W1h = (_Float16*)(ws + W1H_OFF);
    _Float16* W2h = (_Float16*)(ws + W2H_OFF);
    float* WwT = (float*)(ws + WT_OFF);
    float* WbT = (float*)(ws + WT_OFF + WT_BYTES);

    const int transposed = (ws_size >= WT_OFF + 2 * WT_BYTES) ? 1 : 0;

    nnue_convert_w<<<68, 256, 0, stream>>>(W1, W2, W1h, W2h);
    if (transposed) {
        nnue_transpose_w<<<dim3(NFEAT / 32, HACC / 32, 2), dim3(32, 8), 0, stream>>>(
            Ww, Wb, WwT, WbT);
    }
    nnue_row_kernel<<<B_ROWS, 256, 0, stream>>>(
        wfeat, bfeat, stm,
        transposed ? (const float*)WwT : Ww, bw,
        transposed ? (const float*)WbT : Wb, bb,
        act, transposed);
    nnue_mlp_wmma<<<B_ROWS / 16, 32, 0, stream>>>(act, W1h, W2h, b1, b2, Wo, bo, out);
}
